// DiffusionNet_68092411511490
// MI455X (gfx1250) — compile-verified, matrix operand order checked by reading, not run on hardware
//
#include <hip/hip_runtime.h>
#include <cstdint>
#include <cstddef>

// ---------------------------------------------------------------------------
// Problem constants (from reference)
// ---------------------------------------------------------------------------
#define B_SZ   1024
#define DIMN   1024
#define TOK    4
#define NTOK   (B_SZ * TOK)          // 4096 transformer rows
#define HEADS  16
#define DHD    64
#define FFIN   4096                  // FF_INNER
#define EPSLN  1e-5f

// ---------------------------------------------------------------------------
// bf16 helpers
// ---------------------------------------------------------------------------
typedef __attribute__((ext_vector_type(16))) __bf16 v16bf;
typedef __attribute__((ext_vector_type(8)))  float  v8f;

union Frag { v16bf v; unsigned u[8]; };

enum { ACT_NONE = 0, ACT_RELU = 1, ACT_SILU = 2 };

__device__ __forceinline__ unsigned short f2bf(float f) {
  union { float f; unsigned u; } x; x.f = f;
  unsigned r = x.u + 0x7FFFu + ((x.u >> 16) & 1u);   // round-to-nearest-even
  return (unsigned short)(r >> 16);
}

// ---------------------------------------------------------------------------
// Async copy (gfx1250): global -> LDS, 16B per lane, tracked by ASYNCcnt.
// ---------------------------------------------------------------------------
__device__ __forceinline__ void async_ld_b128(unsigned ldsAddr, unsigned gOff,
                                              const void* base) {
  asm volatile("global_load_async_to_lds_b128 %0, %1, %2"
               :: "v"(ldsAddr), "v"(gOff), "s"(base) : "memory");
}
__device__ __forceinline__ void wait_async6() {
  asm volatile("s_wait_asynccnt 0x6" ::: "memory");
}
__device__ __forceinline__ void wait_async0() {
  asm volatile("s_wait_asynccnt 0x0" ::: "memory");
}

// ---------------------------------------------------------------------------
// WMMA bf16 GEMM:  C[M,N] = epilogue( A[M,K] @ W[K,N] )
//  A: bf16 row-major (M x K, pre-converted), Wt: bf16 N-major (N x K).
//  Block tile 128x64, K-step 64, double-buffered LDS fed by async-to-LDS.
//  8 wave32 waves, each owns 32x32 -> 8x v_wmma_f32_16x16x32_bf16 per tile.
// ---------------------------------------------------------------------------
#define BM 128
#define BN 64
#define BK 64
#define LDA_S 72   // bf16 units -> 144B row stride (16B aligned, bank-skewed)
#define LDB_S 72

template<bool HAS_BIAS, int ACT, bool HAS_RES>
__global__ __launch_bounds__(256)
void gemm_kernel(const unsigned short* __restrict__ Ab,   // M x K bf16
                 const unsigned short* __restrict__ Wt,   // N x K bf16
                 const float* __restrict__ bias, const float* __restrict__ Res,
                 float* __restrict__ C, int M, int N, int K)
{
  __shared__ __align__(16) unsigned short sA[2][BM * LDA_S];
  __shared__ __align__(16) unsigned short sB[2][BN * LDB_S];

  const int tid  = threadIdx.x;
  const int lane = tid & 31;
  const int wave = tid >> 5;
  const int wm   = (wave & 3) << 5;           // 0,32,64,96
  const int wn   = (wave >> 2) << 5;          // 0,32
  const int half = lane >> 4;
  const int r16  = lane & 15;

  const int blockM = blockIdx.y * BM;
  const int blockN = blockIdx.x * BN;

  const unsigned sAbase = (unsigned)(size_t)&sA[0][0];
  const unsigned sBbase = (unsigned)(size_t)&sB[0][0];
  const unsigned sAbuf  = (unsigned)sizeof(sA[0]);
  const unsigned sBbuf  = (unsigned)sizeof(sB[0]);

  // issue one tile's async copies: A = 128 rows x 128B (4 instr/wave),
  //                                B =  64 rows x 128B (2 instr/wave)
  auto issue = [&](int buf, int kt) {
#pragma unroll
    for (int i = 0; i < 4; ++i) {
      const int idx = tid + i * 256;
      const int row = idx >> 3, g = idx & 7;            // 8x16B granules per row
      const unsigned lds  = sAbase + buf * sAbuf + (unsigned)((row * LDA_S + g * 8) * 2);
      const unsigned goff = (unsigned)(((size_t)(blockM + row) * K + kt + g * 8) * 2);
      async_ld_b128(lds, goff, Ab);
    }
#pragma unroll
    for (int i = 0; i < 2; ++i) {
      const int idx = tid + i * 256;
      const int row = idx >> 3, g = idx & 7;
      const unsigned lds  = sBbase + buf * sBbuf + (unsigned)((row * LDB_S + g * 8) * 2);
      const unsigned goff = (unsigned)(((size_t)(blockN + row) * K + kt + g * 8) * 2);
      async_ld_b128(lds, goff, Wt);
    }
  };

  v8f acc[2][2];
#pragma unroll
  for (int i = 0; i < 2; ++i)
#pragma unroll
    for (int j = 0; j < 2; ++j)
#pragma unroll
      for (int e = 0; e < 8; ++e) acc[i][j][e] = 0.f;

  // per-lane bf16-pair offsets inside a 16x32 fragment (ISA 7.12.2)
  int koff[8];
#pragma unroll
  for (int j = 0; j < 8; ++j)
    koff[j] = ((j < 4) ? (2 * j) : (16 + 2 * (j - 4))) + 8 * half;

  const int nk = K / BK;
  issue(0, 0);
  for (int t = 0; t < nk; ++t) {
    const int cur = t & 1;
    if (t + 1 < nk) { issue(cur ^ 1, (t + 1) * BK); wait_async6(); }
    else            { wait_async0(); }
    __syncthreads();

    const unsigned short* pa = &sA[cur][0];
    const unsigned short* pb = &sB[cur][0];
#pragma unroll
    for (int ks = 0; ks < 2; ++ks) {                    // two 32-wide k-chunks
      Frag a[2], b[2];
#pragma unroll
      for (int i = 0; i < 2; ++i) {
        const unsigned short* base = pa + (wm + i * 16 + r16) * LDA_S + ks * 32;
#pragma unroll
        for (int j = 0; j < 8; ++j) a[i].u[j] = *(const unsigned*)(base + koff[j]);
      }
#pragma unroll
      for (int i = 0; i < 2; ++i) {
        const unsigned short* base = pb + (wn + i * 16 + r16) * LDB_S + ks * 32;
#pragma unroll
        for (int j = 0; j < 8; ++j) b[i].u[j] = *(const unsigned*)(base + koff[j]);
      }
#pragma unroll
      for (int i = 0; i < 2; ++i)
#pragma unroll
        for (int j = 0; j < 2; ++j)
          acc[i][j] = __builtin_amdgcn_wmma_f32_16x16x32_bf16(
              false, a[i].v, false, b[j].v, (short)0, acc[i][j], false, false);
    }
    __syncthreads();
  }

  // epilogue: D layout: vgpr e -> m = e + 8*half, n = lane&15
#pragma unroll
  for (int i = 0; i < 2; ++i)
#pragma unroll
    for (int j = 0; j < 2; ++j) {
      const int col = blockN + wn + j * 16 + r16;
      float bv = 0.f;
      if (HAS_BIAS) bv = bias[col];
#pragma unroll
      for (int e = 0; e < 8; ++e) {
        const int row = blockM + wm + i * 16 + e + 8 * half;
        float v = acc[i][j][e];
        if (HAS_BIAS) v += bv;
        if (ACT == ACT_RELU) v = fmaxf(v, 0.f);
        if (ACT == ACT_SILU) v = v / (1.f + __expf(-v));
        if (HAS_RES) v += Res[(size_t)row * N + col];
        C[(size_t)row * N + col] = v;
      }
    }
}

// ---------------------------------------------------------------------------
// Weight prep: W[K][N] fp32 -> Wt[N][K] bf16 (tiled transpose, coalesced).
// ---------------------------------------------------------------------------
__global__ __launch_bounds__(256)
void wtrans_kernel(const float* __restrict__ W, unsigned short* __restrict__ Wt,
                   int K, int N)
{
  __shared__ float tile[32][33];
  const int kb = blockIdx.y * 32, nb = blockIdx.x * 32;
  const int tx = threadIdx.x & 31, ty = threadIdx.x >> 5;   // 32 x 8
#pragma unroll
  for (int i = ty; i < 32; i += 8)
    tile[i][tx] = W[(size_t)(kb + i) * N + nb + tx];
  __syncthreads();
#pragma unroll
  for (int i = ty; i < 32; i += 8)
    Wt[(size_t)(nb + i) * K + kb + tx] = f2bf(tile[tx][i]);
}

// Activation prep: fp32 -> bf16 row-major, optional fused pre-ReLU.
template<bool RELU>
__global__ __launch_bounds__(256)
void acvt_kernel(const float* __restrict__ A, unsigned short* __restrict__ Ab, int n4)
{
  int idx = blockIdx.x * blockDim.x + threadIdx.x;   // quads
  if (idx >= n4) return;
  float4 v = ((const float4*)A)[idx];
  if (RELU) {
    v.x = fmaxf(v.x, 0.f); v.y = fmaxf(v.y, 0.f);
    v.z = fmaxf(v.z, 0.f); v.w = fmaxf(v.w, 0.f);
  }
  unsigned lo = (unsigned)f2bf(v.x) | ((unsigned)f2bf(v.y) << 16);
  unsigned hi = (unsigned)f2bf(v.z) | ((unsigned)f2bf(v.w) << 16);
  unsigned* p = (unsigned*)Ab;
  p[2 * idx] = lo;
  p[2 * idx + 1] = hi;
}

// ---------------------------------------------------------------------------
// LayerNorm: out = (x/max? - mean)*rsqrt(var+eps)*g (+ res).  One block per row.
// ---------------------------------------------------------------------------
template<bool STABLE, bool ADD>
__global__ __launch_bounds__(256)
void layernorm_kernel(const float* __restrict__ in, const float* __restrict__ g,
                      const float* __restrict__ res, float* __restrict__ out,
                      int D, long in_stride, long out_stride)
{
  __shared__ float red[256];
  const int row = blockIdx.x;
  const int tid = threadIdx.x;
  const float* x = in + (long)row * in_stride;

  float scale = 1.f;
  if (STABLE) {
    float m = -3.4e38f;
    for (int c = tid; c < D; c += 256) m = fmaxf(m, x[c]);
    red[tid] = m; __syncthreads();
    for (int s = 128; s > 0; s >>= 1) { if (tid < s) red[tid] = fmaxf(red[tid], red[tid + s]); __syncthreads(); }
    scale = 1.f / red[0];
    __syncthreads();
  }
  float sum = 0.f, sq = 0.f;
  for (int c = tid; c < D; c += 256) { float v = x[c] * scale; sum += v; sq += v * v; }
  red[tid] = sum; __syncthreads();
  for (int s = 128; s > 0; s >>= 1) { if (tid < s) red[tid] += red[tid + s]; __syncthreads(); }
  const float mean = red[0] / (float)D;
  __syncthreads();
  red[tid] = sq; __syncthreads();
  for (int s = 128; s > 0; s >>= 1) { if (tid < s) red[tid] += red[tid + s]; __syncthreads(); }
  const float var = red[0] / (float)D - mean * mean;
  const float rstd = rsqrtf(var + EPSLN);

  for (int c = tid; c < D; c += 256) {
    float y = (x[c] * scale - mean) * rstd * g[c];
    if (ADD) y += res[(long)row * out_stride + c];
    out[(long)row * out_stride + c] = y;
  }
}

// ---------------------------------------------------------------------------
// Rotary (ROT_DIM=32) + q-scale.  q rows: NTOK x 1024 (16 heads x 64).
// ---------------------------------------------------------------------------
__global__ void rotary_q_kernel(float* __restrict__ q)
{
  int idx = blockIdx.x * blockDim.x + threadIdx.x;     // NTOK*HEADS
  if (idx >= NTOK * HEADS) return;
  const int row = idx >> 4;
  const int h   = idx & 15;
  const int pos = row & 3;
  float* p = q + (size_t)row * DIMN + h * DHD;
  const float sc = 0.125f;                             // 64^-0.5
#pragma unroll
  for (int d = 32; d < 64; ++d) p[d] *= sc;
#pragma unroll
  for (int m = 0; m < 16; ++m) {
    const float inv = __expf(-0.57564627f * (float)m); // 10000^(-m/16)
    const float ang = (float)pos * inv;
    const float c = __cosf(ang), s = __sinf(ang);
    const float a = p[m] * sc, b = p[m + 16] * sc;
    p[m]      = a * c - b * s;
    p[m + 16] = b * c + a * s;
  }
}

__global__ void rotary_k_kernel(float* __restrict__ kv)  // NTOK x 128, k = cols 0..63
{
  int row = blockIdx.x * blockDim.x + threadIdx.x;
  if (row >= NTOK) return;
  const int pos = row & 3;
  float* p = kv + (size_t)row * 128;
#pragma unroll
  for (int m = 0; m < 16; ++m) {
    const float inv = __expf(-0.57564627f * (float)m);
    const float ang = (float)pos * inv;
    const float c = __cosf(ang), s = __sinf(ang);
    const float a = p[m], b = p[m + 16];
    p[m]      = a * c - b * s;
    p[m + 16] = b * c + a * s;
  }
}

// ---------------------------------------------------------------------------
// Attention core: per (b,h,i): 5-key masked softmax (null kv at j=0).
// rel-pos bucket for n=4 is just max(i+1-j, 0) (< max_exact=16).
// ---------------------------------------------------------------------------
template<bool HAS_BIAS>
__global__ void attn_kernel(const float* __restrict__ q, const float* __restrict__ kv,
                            const float* __restrict__ null_kv, const float* __restrict__ remb,
                            float* __restrict__ out)
{
  int idx = blockIdx.x * blockDim.x + threadIdx.x;     // B*HEADS*TOK
  if (idx >= B_SZ * HEADS * TOK) return;
  const int i = idx & 3;
  const int h = (idx >> 2) & 15;
  const int b = idx >> 6;
  const float* qp = q + ((size_t)(b * 4 + i) * DIMN + h * DHD);

  float s[5] = {0.f, 0.f, 0.f, 0.f, 0.f};
  for (int d = 0; d < DHD; ++d) {
    const float qd = qp[d];
    s[0] += qd * null_kv[d];
#pragma unroll
    for (int j = 1; j < 5; ++j) s[j] += qd * kv[(size_t)(b * 4 + j - 1) * 128 + d];
  }
  if (HAS_BIAS) {
#pragma unroll
    for (int j = 0; j < 5; ++j) {
      int bucket = i + 1 - j; if (bucket < 0) bucket = 0;
      s[j] += remb[bucket * HEADS + h];
    }
  }
  float mx = -3.4e38f;
#pragma unroll
  for (int j = 0; j < 5; ++j) {
    if (j >= i + 2) s[j] = -1e30f;                     // causal (incl. null offset)
    mx = fmaxf(mx, s[j]);
  }
  float p[5], den = 0.f;
#pragma unroll
  for (int j = 0; j < 5; ++j) { p[j] = __expf(s[j] - mx); den += p[j]; }
  const float invd = 1.f / den;

  float* op = out + ((size_t)(b * 4 + i) * DIMN + h * DHD);
  for (int d = 0; d < DHD; ++d) {
    float acc = p[0] * null_kv[DHD + d];
#pragma unroll
    for (int j = 1; j < 5; ++j) acc += p[j] * kv[(size_t)(b * 4 + j - 1) * 128 + DHD + d];
    op[d] = acc * invd;
  }
}

// ---------------------------------------------------------------------------
// Small elementwise kernels
// ---------------------------------------------------------------------------
__global__ void sincos_emb_kernel(const float* __restrict__ t, float* __restrict__ out)
{
  int idx = blockIdx.x * blockDim.x + threadIdx.x;     // B*1024
  if (idx >= B_SZ * DIMN) return;
  const int b = idx >> 10, c = idx & 1023;
  const float e = __expf(-9.2103404f * (float)(c & 511) / 511.f);
  const float a = t[b] * e;
  out[idx] = (c < 512) ? __sinf(a) : __cosf(a);
}

__global__ void stack_kernel(const float* __restrict__ z, const float* __restrict__ te,
                             const float* __restrict__ data, const float* __restrict__ lq,
                             float* __restrict__ x)
{
  int idx = blockIdx.x * blockDim.x + threadIdx.x;     // B*1024
  if (idx >= B_SZ * DIMN) return;
  const int b = idx >> 10, c = idx & 1023;
  float* xr = x + (size_t)b * 4 * DIMN + c;
  xr[0 * DIMN] = z[idx];
  xr[1 * DIMN] = te[idx];
  xr[2 * DIMN] = data[idx];
  xr[3 * DIMN] = lq[c];
}

__global__ void geglu_kernel(const float* __restrict__ h, float* __restrict__ g, int rows)
{
  int idx = blockIdx.x * blockDim.x + threadIdx.x;     // rows*4096
  if (idx >= rows * FFIN) return;
  const int row = idx >> 12, c = idx & 4095;
  const float a  = h[(size_t)row * 2 * FFIN + c];
  const float gt = h[(size_t)row * 2 * FFIN + FFIN + c];
  g[idx] = a * (gt / (1.f + __expf(-gt)));
}

// ---------------------------------------------------------------------------
// Host side
// ---------------------------------------------------------------------------
struct AttnP { const float *norm_g, *wq, *wkv, *null_kv, *wo, *out_norm_g; };
struct FFP   { const float *norm_g, *w1, *w2; };
struct ResP  { const float *w0, *b0, *w1, *b1, *ws; };
struct Net {
  const float *data, *z_hat, *text, *tsteps;
  const float *time_w1, *time_b1, *time_w2, *time_b2;
  ResP zres[3], tres[3];
  const float *learned_query, *rel_pos_emb;
  AttnP selfa[6], crossa[6];
  FFP ff[6];
  const float *out_norm_g, *proj_out;
};

static void fill_attn_ins(const float* const* in, int& k, AttnP& a) {
  a.norm_g = in[k++]; a.wq = in[k++]; a.wkv = in[k++];
  a.null_kv = in[k++]; a.wo = in[k++]; a.out_norm_g = in[k++];
}
static void fill_attn_sorted(const float* const* in, int& k, AttnP& a) {
  a.norm_g = in[k++]; a.null_kv = in[k++]; a.out_norm_g = in[k++];
  a.wkv = in[k++]; a.wo = in[k++]; a.wq = in[k++];
}

static void fill_params_insertion(const float* const* in, int k, Net& p) {
  p.time_w1 = in[k++]; p.time_b1 = in[k++]; p.time_w2 = in[k++]; p.time_b2 = in[k++];
  for (int i = 0; i < 3; ++i) {
    p.zres[i].w0 = in[k++]; p.zres[i].b0 = in[k++];
    p.zres[i].w1 = in[k++]; p.zres[i].b1 = in[k++];
    p.zres[i].ws = (i == 0) ? in[k++] : nullptr;
  }
  for (int i = 0; i < 3; ++i) {
    p.tres[i].w0 = in[k++]; p.tres[i].b0 = in[k++];
    p.tres[i].w1 = in[k++]; p.tres[i].b1 = in[k++];
    p.tres[i].ws = (i == 0) ? in[k++] : nullptr;
  }
  p.learned_query = in[k++]; p.rel_pos_emb = in[k++];
  for (int l = 0; l < 6; ++l) {
    fill_attn_ins(in, k, p.selfa[l]);
    fill_attn_ins(in, k, p.crossa[l]);
    p.ff[l].norm_g = in[k++]; p.ff[l].w1 = in[k++]; p.ff[l].w2 = in[k++];
  }
  p.out_norm_g = in[k++]; p.proj_out = in[k++];
}

static void fill_params_sorted(const float* const* in, int k, Net& p) {
  for (int l = 0; l < 6; ++l) {                // keys sorted: cross, ff, self
    fill_attn_sorted(in, k, p.crossa[l]);
    p.ff[l].norm_g = in[k++]; p.ff[l].w1 = in[k++]; p.ff[l].w2 = in[k++];
    fill_attn_sorted(in, k, p.selfa[l]);
  }
  p.learned_query = in[k++]; p.out_norm_g = in[k++];
  p.proj_out = in[k++];      p.rel_pos_emb = in[k++];
  for (int i = 0; i < 3; ++i) {                // text_resnet: b0,b1,w0,w1,(ws)
    p.tres[i].b0 = in[k++]; p.tres[i].b1 = in[k++];
    p.tres[i].w0 = in[k++]; p.tres[i].w1 = in[k++];
    p.tres[i].ws = (i == 0) ? in[k++] : nullptr;
  }
  p.time_b1 = in[k++]; p.time_b2 = in[k++]; p.time_w1 = in[k++]; p.time_w2 = in[k++];
  for (int i = 0; i < 3; ++i) {
    p.zres[i].b0 = in[k++]; p.zres[i].b1 = in[k++];
    p.zres[i].w0 = in[k++]; p.zres[i].w1 = in[k++];
    p.zres[i].ws = (i == 0) ? in[k++] : nullptr;
  }
}

// One GEMM = activation bf16 convert + weight transpose/convert + WMMA kernel.
static void gemm2(hipStream_t s, const float* A, const float* W,
                  const float* bias, const float* Res, float* C,
                  int M, int N, int K, bool preRelu, int act,
                  unsigned short* aScr, unsigned short* wScr)
{
  {
    const int n4 = (M * K) / 4;
    dim3 g((n4 + 255) / 256);
    if (preRelu) acvt_kernel<true ><<<g, 256, 0, s>>>(A, aScr, n4);
    else         acvt_kernel<false><<<g, 256, 0, s>>>(A, aScr, n4);
  }
  wtrans_kernel<<<dim3(N / 32, K / 32), 256, 0, s>>>(W, wScr, K, N);

  dim3 grid(N / BN, M / BM), blk(256);
  if (act == ACT_SILU) {
    gemm_kernel<true, ACT_SILU, false><<<grid, blk, 0, s>>>(aScr, wScr, bias, Res, C, M, N, K);
  } else if (bias && Res) {
    gemm_kernel<true, ACT_NONE, true ><<<grid, blk, 0, s>>>(aScr, wScr, bias, Res, C, M, N, K);
  } else if (bias) {
    gemm_kernel<true, ACT_NONE, false><<<grid, blk, 0, s>>>(aScr, wScr, bias, Res, C, M, N, K);
  } else if (Res) {
    gemm_kernel<false, ACT_NONE, true ><<<grid, blk, 0, s>>>(aScr, wScr, bias, Res, C, M, N, K);
  } else {
    gemm_kernel<false, ACT_NONE, false><<<grid, blk, 0, s>>>(aScr, wScr, bias, Res, C, M, N, K);
  }
}

extern "C" void kernel_launch(void* const* d_in, const int* in_sizes, int n_in,
                              void* d_out, int out_size, void* d_ws, size_t ws_size,
                              hipStream_t stream)
{
  const float* const* in = (const float* const*)d_in;
  Net P;

  // Detect flattening order of the params pytree from in_sizes.
  if (n_in >= 2 && in_sizes[1] == B_SZ) {
    // fully sorted pytree: data, diffusion_timesteps, <params sorted>, text, z_hat
    P.data = in[0]; P.tsteps = in[1];
    fill_params_sorted(in, 2, P);
    P.text = in[n_in - 2]; P.z_hat = in[n_in - 1];
  } else {
    // signature order: data, z_hat, text, diffusion_timesteps, <params>
    P.data = in[0]; P.z_hat = in[1]; P.text = in[2]; P.tsteps = in[3];
    if (n_in > 4 && in_sizes[4] == DIMN) fill_params_sorted(in, 4, P);     // sorted leaves
    else                                  fill_params_insertion(in, 4, P); // insertion order
  }

  // ---- workspace carve (bump allocator, 256B aligned) ----
  char* wsp = (char*)d_ws;
  size_t off = 0;
  auto allocB = [&](size_t bytes) -> void* {
    void* p = (void*)(wsp + off);
    off += (bytes + 255) & ~((size_t)255);
    return p;
  };
  auto alloc = [&](size_t nfloats) -> float* { return (float*)allocB(nfloats * 4); };

  float* tA   = alloc((size_t)B_SZ * 4096);
  float* tB   = alloc((size_t)B_SZ * 4096);
  float* tC   = alloc((size_t)B_SZ * 4096);     // final context lives here
  float* zA   = alloc((size_t)B_SZ * DIMN);
  float* zB   = alloc((size_t)B_SZ * DIMN);
  float* zC   = alloc((size_t)B_SZ * DIMN);
  float* x    = alloc((size_t)NTOK * DIMN);
  float* xn   = alloc((size_t)NTOK * DIMN);
  float* qbuf = alloc((size_t)NTOK * DIMN);
  float* pbuf = alloc((size_t)NTOK * DIMN);
  float* kvb  = alloc((size_t)NTOK * 128);
  float* hbuf = alloc((size_t)B_SZ * 2 * FFIN); // FF hidden, one M-chunk of 1024 rows
  float* gbuf = alloc((size_t)B_SZ * FFIN);
  unsigned short* aScr = (unsigned short*)allocB((size_t)4096 * 1024 * 2); // max MxK bf16
  unsigned short* wScr = (unsigned short*)allocB((size_t)4096 * 4096 * 2); // max KxN bf16
  float* aob  = xn;                              // alias: xn dead once attn core ran
  float* fbuf = zA;                              // alias: free after stack

  const dim3 b256(256);
  auto g1d = [](size_t n) { return dim3((unsigned)((n + 255) / 256)); };

  // ================= text resnet -> context (B x 4 x 1024 in tC) =================
  gemm2(stream, P.text, P.tres[0].w0, P.tres[0].b0, nullptr, tA, B_SZ, 768, 768, true, ACT_NONE, aScr, wScr);
  gemm2(stream, tA, P.tres[0].w1, P.tres[0].b1, nullptr, tB, B_SZ, 4096, 768, true, ACT_NONE, aScr, wScr);
  gemm2(stream, P.text, P.tres[0].ws, nullptr, tB, tC, B_SZ, 4096, 768, false, ACT_NONE, aScr, wScr);
  gemm2(stream, tC, P.tres[1].w0, P.tres[1].b0, nullptr, tA, B_SZ, 4096, 4096, true, ACT_NONE, aScr, wScr);
  gemm2(stream, tA, P.tres[1].w1, P.tres[1].b1, tC, tB, B_SZ, 4096, 4096, true, ACT_NONE, aScr, wScr);
  gemm2(stream, tB, P.tres[2].w0, P.tres[2].b0, nullptr, tA, B_SZ, 4096, 4096, true, ACT_NONE, aScr, wScr);
  gemm2(stream, tA, P.tres[2].w1, P.tres[2].b1, tB, tC, B_SZ, 4096, 4096, true, ACT_NONE, aScr, wScr);

  // ================= z resnet -> zC =================
  gemm2(stream, P.z_hat, P.zres[0].w0, P.zres[0].b0, nullptr, zA, B_SZ, 512, 512, true, ACT_NONE, aScr, wScr);
  gemm2(stream, zA, P.zres[0].w1, P.zres[0].b1, nullptr, zB, B_SZ, DIMN, 512, true, ACT_NONE, aScr, wScr);
  gemm2(stream, P.z_hat, P.zres[0].ws, nullptr, zB, zC, B_SZ, DIMN, 512, false, ACT_NONE, aScr, wScr);
  gemm2(stream, zC, P.zres[1].w0, P.zres[1].b0, nullptr, zA, B_SZ, DIMN, DIMN, true, ACT_NONE, aScr, wScr);
  gemm2(stream, zA, P.zres[1].w1, P.zres[1].b1, zC, zB, B_SZ, DIMN, DIMN, true, ACT_NONE, aScr, wScr);
  gemm2(stream, zB, P.zres[2].w0, P.zres[2].b0, nullptr, zA, B_SZ, DIMN, DIMN, true, ACT_NONE, aScr, wScr);
  gemm2(stream, zA, P.zres[2].w1, P.zres[2].b1, zB, zC, B_SZ, DIMN, DIMN, true, ACT_NONE, aScr, wScr);

  // ================= time MLP (reuse zA/zB) =================
  float* teA = zA; float* teB = zB;
  sincos_emb_kernel<<<g1d((size_t)B_SZ * DIMN), b256, 0, stream>>>(P.tsteps, teA);
  gemm2(stream, teA, P.time_w1, P.time_b1, nullptr, teB, B_SZ, DIMN, DIMN, false, ACT_SILU, aScr, wScr);
  gemm2(stream, teB, P.time_w2, P.time_b2, nullptr, teA, B_SZ, DIMN, DIMN, false, ACT_NONE, aScr, wScr);

  // ================= stack x = [z, te, data, lq] =================
  stack_kernel<<<g1d((size_t)B_SZ * DIMN), b256, 0, stream>>>(zC, teA, P.data, P.learned_query, x);

  // ================= transformer layers =================
  for (int l = 0; l < 6; ++l) {
    // ---- self attention (with rel-pos bias) ----
    {
      const AttnP& a = P.selfa[l];
      layernorm_kernel<false, false><<<NTOK, b256, 0, stream>>>(x, a.norm_g, nullptr, xn, DIMN, DIMN, DIMN);
      gemm2(stream, xn, a.wq, nullptr, nullptr, qbuf, NTOK, DIMN, DIMN, false, ACT_NONE, aScr, wScr);
      gemm2(stream, xn, a.wkv, nullptr, nullptr, kvb, NTOK, 128, DIMN, false, ACT_NONE, aScr, wScr);
      rotary_q_kernel<<<g1d((size_t)NTOK * HEADS), b256, 0, stream>>>(qbuf);
      rotary_k_kernel<<<g1d(NTOK), b256, 0, stream>>>(kvb);
      attn_kernel<true><<<g1d((size_t)B_SZ * HEADS * TOK), b256, 0, stream>>>(qbuf, kvb, a.null_kv, P.rel_pos_emb, aob);
      gemm2(stream, aob, a.wo, nullptr, nullptr, pbuf, NTOK, DIMN, DIMN, false, ACT_NONE, aScr, wScr);
      layernorm_kernel<false, true><<<NTOK, b256, 0, stream>>>(pbuf, a.out_norm_g, x, x, DIMN, DIMN, DIMN);
    }
    // ---- cross attention (context kv, no bias) ----
    {
      const AttnP& a = P.crossa[l];
      layernorm_kernel<false, false><<<NTOK, b256, 0, stream>>>(x, a.norm_g, nullptr, xn, DIMN, DIMN, DIMN);
      gemm2(stream, xn, a.wq, nullptr, nullptr, qbuf, NTOK, DIMN, DIMN, false, ACT_NONE, aScr, wScr);
      gemm2(stream, tC, a.wkv, nullptr, nullptr, kvb, NTOK, 128, DIMN, false, ACT_NONE, aScr, wScr);
      rotary_q_kernel<<<g1d((size_t)NTOK * HEADS), b256, 0, stream>>>(qbuf);
      rotary_k_kernel<<<g1d(NTOK), b256, 0, stream>>>(kvb);
      attn_kernel<false><<<g1d((size_t)B_SZ * HEADS * TOK), b256, 0, stream>>>(qbuf, kvb, a.null_kv, P.rel_pos_emb, aob);
      gemm2(stream, aob, a.wo, nullptr, nullptr, pbuf, NTOK, DIMN, DIMN, false, ACT_NONE, aScr, wScr);
      layernorm_kernel<false, true><<<NTOK, b256, 0, stream>>>(pbuf, a.out_norm_g, x, x, DIMN, DIMN, DIMN);
    }
    // ---- GEGLU feed-forward, chunked over M to bound workspace ----
    {
      const FFP& f = P.ff[l];
      layernorm_kernel<false, false><<<NTOK, b256, 0, stream>>>(x, f.norm_g, nullptr, xn, DIMN, DIMN, DIMN);
      for (int c = 0; c < 4; ++c) {
        const size_t ro = (size_t)c * 1024;
        gemm2(stream, xn + ro * DIMN, f.w1, nullptr, nullptr, hbuf, 1024, 2 * FFIN, DIMN, false, ACT_NONE, aScr, wScr);
        geglu_kernel<<<g1d((size_t)1024 * FFIN), b256, 0, stream>>>(hbuf, gbuf, 1024);
        gemm2(stream, gbuf, f.w2, nullptr, x + ro * DIMN, x + ro * DIMN, 1024, DIMN, FFIN, false, ACT_NONE, aScr, wScr);
      }
    }
  }

  // ================= final: stable LN of token 3, project =================
  layernorm_kernel<true, false><<<B_SZ, b256, 0, stream>>>(
      x + 3 * DIMN, P.out_norm_g, nullptr, fbuf, DIMN, (long)TOK * DIMN, DIMN);
  gemm2(stream, fbuf, P.proj_out, nullptr, nullptr, (float*)d_out, B_SZ, DIMN, DIMN, false, ACT_NONE, aScr, wScr);

  (void)in_sizes; (void)n_in; (void)out_size; (void)ws_size;
}
